// DeepsetGNN_42210938585863
// MI455X (gfx1250) — compile-verified, matrix operand order checked by reading, not run on hardware
//
#include <hip/hip_runtime.h>

// ---------------------------------------------------------------------------
// DeepSet GNN for MI455X (gfx1250): wave32 + v_wmma_f32_16x16x32_f16
// + async global->LDS weight staging (ASYNCcnt path) when available.
// ---------------------------------------------------------------------------

typedef __attribute__((ext_vector_type(16))) _Float16 v16h;
typedef __attribute__((ext_vector_type(8)))  float    v8f;
typedef int v4i __attribute__((vector_size(16)));

#define NPART 8192
#define KNBR  32
#define HDIM  128

union HF { v16h v; unsigned long long q[4]; };

__device__ __forceinline__ float gelu_f(float x) {
    // jax.nn.gelu default (tanh approximation)
    const float k0 = 0.7978845608028654f;
    const float k1 = 0.044715f;
    float u = k0 * (x + k1 * x * x * x);
    return 0.5f * x * (1.0f + tanhf(u));
}

// Load one 16x32 f16 A/B fragment (ISA 7.12.2 16-bit layout) as 4x b64
// (compiler merges into 2x b128). base = &buf[row*128 + ks*32 + hi8].
__device__ __forceinline__ v16h load_frag(const _Float16* base) {
    HF f;
    f.q[0] = *(const unsigned long long*)(base + 0);   // K {0..3}
    f.q[1] = *(const unsigned long long*)(base + 4);   // K {4..7}
    f.q[2] = *(const unsigned long long*)(base + 16);  // K {16..19}
    f.q[3] = *(const unsigned long long*)(base + 20);  // K {20..23}
    return f.v;
}

// One wave computes a 16-row x (NCT*16)-col strip with K=128, f32 accumulate.
// sH: 64x128 f16 activations.  sW: 128x128 f16 weights TRANSPOSED (W[n][k]).
template<int NCT>
__device__ __forceinline__ void gemm_tile(const _Float16* sH, const _Float16* sW,
                                          int rowBase, int ctBase, int lane,
                                          v8f acc[NCT]) {
    const int lrow = lane & 15;
    const int hi8  = (lane & 16) ? 8 : 0;
    v8f z = {0.f, 0.f, 0.f, 0.f, 0.f, 0.f, 0.f, 0.f};
#pragma unroll
    for (int ct = 0; ct < NCT; ++ct) acc[ct] = z;
#pragma unroll
    for (int ks = 0; ks < 4; ++ks) {
        v16h a = load_frag(sH + (rowBase + lrow) * HDIM + ks * 32 + hi8);
#pragma unroll
        for (int ct = 0; ct < NCT; ++ct) {
            v16h b = load_frag(sW + ((ctBase + ct) * 16 + lrow) * HDIM + ks * 32 + hi8);
            acc[ct] = __builtin_amdgcn_wmma_f32_16x16x32_f16(
                false, a, false, b, (short)0, acc[ct], false, false);
        }
    }
}

// Write gelu(acc + bias) back as f16 into the wave's own rows/cols of sH.
template<int NCT>
__device__ __forceinline__ void store_act(_Float16* sH, int rowBase, int ctBase,
                                          int lane, v8f acc[NCT], const float* bias) {
    const int lrow = lane & 15;
    const int hi8  = (lane & 16) ? 8 : 0;
#pragma unroll
    for (int ct = 0; ct < NCT; ++ct) {
        int col = (ctBase + ct) * 16 + lrow;
        float bb = bias ? bias[col] : 0.f;
#pragma unroll
        for (int r = 0; r < 8; ++r) {
            int row = rowBase + r + hi8;
            sH[row * HDIM + col] = (_Float16)gelu_f(acc[ct][r] + bb);
        }
    }
}

// ---- async global->LDS staging of one 128x128 f16 weight matrix (32 KB) ----
#if __has_builtin(__builtin_amdgcn_global_load_async_to_lds_b128)
#define USE_ASYNC_LDS 1
#else
#define USE_ASYNC_LDS 0
#endif

__device__ __forceinline__ void copy_w(const _Float16* __restrict__ src,
                                       _Float16* dst, int t) {
#if USE_ASYNC_LDS
    for (int e = t; e < (HDIM * HDIM) / 8; e += 256) {     // 2048 x 16B
        __builtin_amdgcn_global_load_async_to_lds_b128(
            (__attribute__((address_space(1))) v4i*)(src + e * 8),
            (__attribute__((address_space(3))) v4i*)(dst + e * 8),
            0, 0);
    }
#else
    for (int e = t; e < (HDIM * HDIM) / 8; e += 256) {
        ((uint4*)dst)[e] = ((const uint4*)src)[e];
    }
#endif
}

__device__ __forceinline__ void wait_copy_w() {
#if USE_ASYNC_LDS
# if __has_builtin(__builtin_amdgcn_s_wait_asynccnt)
    __builtin_amdgcn_s_wait_asynccnt(0);
# else
    asm volatile("s_wait_asynccnt 0" ::: "memory");
# endif
#endif
}

// --------------------------- K0: weight prep -------------------------------
// Transpose + f32->f16 convert W1,W2,D0,D1 into Wt[n][k] layout.
__global__ __launch_bounds__(256) void prep_kernel(
    const float* __restrict__ W1, const float* __restrict__ W2,
    const float* __restrict__ D0, const float* __restrict__ D1,
    _Float16* __restrict__ dst) {
    int id = blockIdx.x * 256 + threadIdx.x;   // 16384 threads
    for (int e = id; e < 4 * HDIM * HDIM; e += 64 * 256) {
        int m = e >> 14;
        int rem = e & 16383;
        int k = rem >> 7, n = rem & 127;
        const float* src = (m == 0) ? W1 : (m == 1) ? W2 : (m == 2) ? D0 : D1;
        dst[m * HDIM * HDIM + n * HDIM + k] = (_Float16)src[k * HDIM + n];
    }
}

// --------------------------- K1: neighbor search ---------------------------
// One block per particle. Deterministic top-32 by repeated min extraction of
// packed (dist_bits<<32 | j) keys (tie-break by lower index, like top_k).
__global__ __launch_bounds__(256) void nn_kernel(const float* __restrict__ xs,
                                                 int* __restrict__ inds) {
    __shared__ unsigned sD[NPART];             // 32 KB of dist2 bit patterns
    __shared__ unsigned long long sMin[256];
    __shared__ unsigned long long sSel;
    const int t = threadIdx.x;
    const int i = blockIdx.x;
    const float xi0 = xs[2 * i], xi1 = xs[2 * i + 1];

    unsigned long long m = ~0ull;
    for (int c = 0; c < 32; ++c) {
        int j = t + (c << 8);
        float d0 = xi0 - xs[2 * j];     d0 -= rintf(d0);   // min-image, width=1
        float d1 = xi1 - xs[2 * j + 1]; d1 -= rintf(d1);
        float d2 = d0 * d0 + d1 * d1;
        unsigned bits = __float_as_uint(d2);               // monotone for d2>=0
        sD[j] = bits;
        unsigned long long key = ((unsigned long long)bits << 32) | (unsigned)j;
        m = (key < m) ? key : m;
    }
    sMin[t] = m;
    __syncthreads();

    for (int r = 0; r < KNBR; ++r) {
        if (t == 0) {
            unsigned long long best = sMin[0];
            for (int u = 1; u < 256; ++u) if (sMin[u] < best) best = sMin[u];
            sSel = best;
            inds[i * KNBR + r] = (int)(best & 0xffffffffu);
        }
        __syncthreads();
        unsigned jw = (unsigned)(sSel & 0xffffffffu);
        if ((jw & 255u) == (unsigned)t) {
            sD[jw] = 0xffffffffu;                          // remove winner
            unsigned long long mm = ~0ull;
            for (int c = 0; c < 32; ++c) {
                int j = t + (c << 8);
                unsigned long long key =
                    ((unsigned long long)sD[j] << 32) | (unsigned)j;
                mm = (key < mm) ? key : mm;
            }
            sMin[t] = mm;
        }
        __syncthreads();
    }
}

// --------------------------- K2: encoder + pool ----------------------------
// Block = 2 particles (64 edge rows). 8 waves = 4 row-tiles x 2 col-halves.
// Weights staged (async) into LDS once per layer per block.
__global__ __launch_bounds__(256) void enc_pool_kernel(
    const float* __restrict__ xs, const float* __restrict__ gs,
    const int* __restrict__ inds,
    const float* __restrict__ W0, const float* __restrict__ b0,
    const _Float16* __restrict__ W1t, const float* __restrict__ b1,
    const _Float16* __restrict__ W2t, const float* __restrict__ b2,
    float* __restrict__ pool) {
    __shared__ _Float16 sH[64 * HDIM];         // 16 KB activations
    __shared__ _Float16 sW[HDIM * HDIM];       // 32 KB current weights (Wt)
    __shared__ float sInp[64 * 4];
    __shared__ float sPart[8 * 32 * 4];        // [wave][lane][ct] pool partials
    const int t = threadIdx.x;
    const int p0 = blockIdx.x * 2;

    copy_w(W1t, sW, t);                        // kick off weight DMA first

    if (t < 64) {
        int p = p0 + (t >> 5);
        int k = t & 31;
        int j = inds[p * KNBR + k];
        float d0 = xs[2 * p] - xs[2 * j];         d0 -= rintf(d0);
        float d1 = xs[2 * p + 1] - xs[2 * j + 1]; d1 -= rintf(d1);
        sInp[t * 4 + 0] = d0;
        sInp[t * 4 + 1] = d1;
        sInp[t * 4 + 2] = gs[2 * p] - gs[2 * j];
        sInp[t * 4 + 3] = gs[2 * p + 1] - gs[2 * j + 1];
    }
    __syncthreads();

    // layer 0 (VALU): h0 = gelu(inp @ W0 + b0) -> f16 tile (overlaps weight DMA)
    for (int e = t; e < 64 * HDIM; e += 256) {
        int row = e >> 7, col = e & 127;
        float a = b0[col];
        a += sInp[row * 4 + 0] * W0[0 * HDIM + col];
        a += sInp[row * 4 + 1] * W0[1 * HDIM + col];
        a += sInp[row * 4 + 2] * W0[2 * HDIM + col];
        a += sInp[row * 4 + 3] * W0[3 * HDIM + col];
        sH[e] = (_Float16)gelu_f(a);
    }
    wait_copy_w();
    __syncthreads();                           // sH + sW(W1t) ready

    const int wave = t >> 5, lane = t & 31;
    const int rowBase = (wave >> 1) * 16;      // 4 row-tiles
    const int ctBase  = (wave & 1) * 4;        // 2 column halves
    v8f acc[4];

    gemm_tile<4>(sH, sW, rowBase, ctBase, lane, acc);     // layer 1
    store_act<4>(sH, rowBase, ctBase, lane, acc, b1);
    __syncthreads();                           // L1 done; sW reads done

    copy_w(W2t, sW, t);
    wait_copy_w();
    __syncthreads();                           // sW(W2t) ready

    gemm_tile<4>(sH, sW, rowBase, ctBase, lane, acc);     // layer 2 (linear)

    // partial pool: sum this lane's 8 rows per column tile (deterministic)
#pragma unroll
    for (int ct = 0; ct < 4; ++ct) {
        float s = 0.f;
#pragma unroll
        for (int r = 0; r < 8; ++r) s += acc[ct][r];
        sPart[(wave * 32 + lane) * 4 + ct] = s;
    }
    __syncthreads();

    // combine in fixed order; fold K*b2 into pool
    for (int e = t; e < 2 * HDIM; e += 256) {
        int p = e >> 7, col = e & 127;
        int ct = col >> 4, lrow = col & 15;
        float s = 32.0f * b2[col];
#pragma unroll
        for (int rt = 0; rt < 2; ++rt) {           // row-tiles 2p, 2p+1
            int w = 2 * (2 * p + rt) + (ct >> 2);  // wave owning (rt, ct-half)
            s += sPart[(w * 32 + lrow) * 4 + (ct & 3)];
            s += sPart[(w * 32 + 16 + lrow) * 4 + (ct & 3)];
        }
        pool[(p0 + p) * HDIM + col] = s;
    }
}

// --------------------------- K3: decoder -----------------------------------
__global__ __launch_bounds__(256) void dec_kernel(
    const float* __restrict__ pool,
    const _Float16* __restrict__ D0t, const _Float16* __restrict__ D1t,
    const float* __restrict__ D2, float* __restrict__ out) {
    __shared__ _Float16 sH[64 * HDIM];         // 16 KB
    __shared__ _Float16 sW[HDIM * HDIM];       // 32 KB
    const int t = threadIdx.x;
    const int r0 = blockIdx.x * 64;

    copy_w(D0t, sW, t);
    for (int e = t; e < 64 * HDIM; e += 256)
        sH[e] = (_Float16)pool[r0 * HDIM + e];
    wait_copy_w();
    __syncthreads();

    const int wave = t >> 5, lane = t & 31;
    const int rowBase = (wave >> 1) * 16;
    const int ctBase  = (wave & 1) * 4;
    v8f acc[4];

    gemm_tile<4>(sH, sW, rowBase, ctBase, lane, acc);
    store_act<4>(sH, rowBase, ctBase, lane, acc, nullptr);
    __syncthreads();

    copy_w(D1t, sW, t);
    wait_copy_w();
    __syncthreads();

    gemm_tile<4>(sH, sW, rowBase, ctBase, lane, acc);
    store_act<4>(sH, rowBase, ctBase, lane, acc, nullptr);
    __syncthreads();

    // final 128 -> 2 (no bias): 64 rows x 2 outputs
    if (t < 128) {
        int row = t >> 1, d = t & 1;
        float s = 0.f;
        for (int n = 0; n < HDIM; ++n)
            s += (float)sH[row * HDIM + n] * D2[n * 2 + d];
        out[(r0 + row) * 2 + d] = s;
    }
}

// ---------------------------------------------------------------------------
extern "C" void kernel_launch(void* const* d_in, const int* in_sizes, int n_in,
                              void* d_out, int out_size, void* d_ws, size_t ws_size,
                              hipStream_t stream) {
    const float* xs = (const float*)d_in[0];
    const float* gs = (const float*)d_in[1];
    const float* W0 = (const float*)d_in[2];
    const float* b0 = (const float*)d_in[3];
    const float* W1 = (const float*)d_in[4];
    const float* b1 = (const float*)d_in[5];
    const float* W2 = (const float*)d_in[6];
    const float* b2 = (const float*)d_in[7];
    const float* D0 = (const float*)d_in[8];
    const float* D1 = (const float*)d_in[9];
    const float* D2 = (const float*)d_in[10];
    float* out = (float*)d_out;

    char* ws = (char*)d_ws;
    int*      inds = (int*)ws;                                    // 1 MB
    float*    pool = (float*)(ws + (size_t)(1 << 20));            // 4 MB
    _Float16* Wt   = (_Float16*)(ws + (size_t)5 * (1 << 20));     // 4x 32 KB
    _Float16* W1t = Wt;
    _Float16* W2t = Wt + 1 * HDIM * HDIM;
    _Float16* D0t = Wt + 2 * HDIM * HDIM;
    _Float16* D1t = Wt + 3 * HDIM * HDIM;

    prep_kernel<<<64, 256, 0, stream>>>(W1, W2, D0, D1, Wt);
    nn_kernel<<<NPART, 256, 0, stream>>>(xs, inds);
    enc_pool_kernel<<<NPART / 2, 256, 0, stream>>>(xs, gs, inds, W0, b0,
                                                   W1t, b1, W2t, b2, pool);
    dec_kernel<<<NPART / 64, 256, 0, stream>>>(pool, D0t, D1t, D2, out);
}
